// Model_81973745811845
// MI455X (gfx1250) — compile-verified, matrix operand order checked by reading, not run on hardware
//
#include <hip/hip_runtime.h>

// ---------------- types for WMMA ----------------
typedef _Float16 v16h __attribute__((ext_vector_type(16)));
typedef float    v8f  __attribute__((ext_vector_type(8)));

union HF { v16h v; unsigned int u[8]; };

// gfx1250 async global->LDS copy (ASYNCcnt-tracked), with safe fallback.
#if defined(__has_builtin)
#if __has_builtin(__builtin_amdgcn_global_load_async_to_lds_b128)
#define HAVE_ASYNC_LDS 1
#endif
#endif
#ifndef HAVE_ASYNC_LDS
#define HAVE_ASYNC_LDS 0
#endif

// 128-bit vector type matching the builtin's V4i pointee, with AS variants.
typedef int int4v __attribute__((__vector_size__(16)));
#if HAVE_ASYNC_LDS
typedef __attribute__((address_space(1))) int4v g4v;   // global (AS1)
typedef __attribute__((address_space(3))) int4v l4v;   // LDS (AS3)
#endif

__device__ __forceinline__ void copy16_to_lds(const void* g, void* l) {
#if HAVE_ASYNC_LDS
  void* gv = const_cast<void*>(g);
  __builtin_amdgcn_global_load_async_to_lds_b128((g4v*)gv, (l4v*)l, 0, 0);
#else
  *(uint4*)l = *(const uint4*)g;
#endif
}
__device__ __forceinline__ void copy_join() {
#if HAVE_ASYNC_LDS
#if __has_builtin(__builtin_amdgcn_s_wait_asynccnt)
  __builtin_amdgcn_s_wait_asynccnt(0);
#else
  asm volatile("s_wait_asynccnt 0" ::: "memory");
#endif
#endif
}

// ---------------- problem dims ----------------
constexpr int Bb   = 16;
constexpr int Lctx = 512;
constexpr int Nvar = 512;
constexpr int MARK = 4;
constexpr int Dm   = 512;
constexpr int Hh   = 8;
constexpr int DFF  = 2048;
constexpr int NLay = 2;
constexpr int Pp   = 336;
constexpr int Mm   = 512;
constexpr int Kk   = 10;
constexpr int DH   = 64;
constexpr int Tt   = Nvar + MARK;   // 516 tokens
constexpr int Tpad = 528;           // 33*16 (and a multiple of 8)
constexpr int BT   = Bb * Tpad;     // 8448 padded rows
constexpr int PN   = Pp * Nvar;     // 172032
constexpr int MB   = Mm + Bb;       // 528 bank rows
constexpr int nQT  = Tpad / 16;     // 33 q-tiles
constexpr int OPB  = Tpad / 8;      // 66 row-octets per batch

// GEMM tiling
constexpr int BM = 64, BN = 128, BK = 32;
constexpr int SA  = BK + 8;          // LDS stride for A (halfs) = 40
constexpr int SBD = 20;              // LDS stride for B (dwords per column)

__device__ __forceinline__ v8f wmma16(v16h a, v16h b, v8f c) {
  // v_wmma_f32_16x16x32_f16: (neg_a, A, neg_b, B, c_mod, C, reuse_a, reuse_b)
  return __builtin_amdgcn_wmma_f32_16x16x32_f16(false, a, false, b, (short)0, c,
                                                false, false);
}

__device__ __forceinline__ v8f zero8() {
  v8f z;
#pragma unroll
  for (int r = 0; r < 8; ++r) z[r] = 0.f;
  return z;
}

// ISA 7.12.2 fragment recap (16-bit A 16x32, lane l, hi = l>>4):
//   halves  [8*hi, 8*hi+8)  and  [16+8*hi, 24+8*hi)   -> two 16-B runs.
// B mirrors with lane&15 selecting the column; with operands stored
// octet-interleaved ([k/8][n][8 halves]) the B dwords also form two 16-B runs.
__device__ __forceinline__ float gelu_t(float x) {
  float x3 = x * x * x;
  return 0.5f * x * (1.f + tanhf(0.7978845608f * (x + 0.044715f * x3)));
}

// ---------------- small helper kernels ----------------
// f32 -> f16 with row-OCTET interleave:
//   dst[((k/8)*Nc + n)*8 + ((k>>1)&3)*2 + (k&1)] = src[k*Nc + n]
// so each (n, k-octet) cell is 16 contiguous bytes holding k..k+7 at column n.
__global__ void cvt_f32_f16_oct(const float* __restrict__ s,
                                _Float16* __restrict__ d, int Kd, int Nc) {
  int i = blockIdx.x * blockDim.x + threadIdx.x;
  if (i >= Kd * Nc) return;
  int k = i / Nc, n = i - k * Nc;
  d[(((size_t)(k >> 3)) * Nc + n) * 8 + ((k >> 1) & 3) * 2 + (k & 1)] =
      (_Float16)s[i];
}

// Build inverted-embedding token matrix [B*Tpad, L] (f16), pad rows zero.
__global__ void pack_tok(const float* __restrict__ x_enc,
                         const float* __restrict__ x_mark,
                         _Float16* __restrict__ tok) {
  int i = blockIdx.x * blockDim.x + threadIdx.x;
  if (i >= BT * Lctx) return;
  int l   = i & (Lctx - 1);
  int row = i >> 9;
  int t   = row % Tpad;
  int b   = row / Tpad;
  float v = 0.f;
  if (t < Nvar)      v = x_enc[((size_t)b * Lctx + l) * Nvar + t];
  else if (t < Tt)   v = x_mark[((size_t)b * Lctx + l) * MARK + (t - Nvar)];
  tok[i] = (_Float16)v;
}

// ---------------- WMMA GEMM ----------------
// C = A(f16)[Mr,Kd] @ B + bias.  B is PRE-SWIZZLED into octet cells
// ([Kd/8][Nc][8]).  Staging is branchless (address-clamped; clamped data only
// feeds store-masked outputs) and goes through global_load_async_to_lds_b128.
// swzC=1 stores the f16 output octet-interleaved over rows (used for V).
__global__ __launch_bounds__(256) void gemm_wmma(
    const _Float16* __restrict__ A, const _Float16* __restrict__ Bw,
    const float* __restrict__ bias, float* __restrict__ Cf,
    _Float16* __restrict__ Ch, int Mr, int Kd, int Nc, int act, int swzC) {
  __shared__ alignas(16) _Float16 sA[BM * SA];
  __shared__ alignas(16) unsigned int sBd[BN * SBD];
  int nt = (Nc + BN - 1) / BN;
  int bm = blockIdx.x / nt, bn = blockIdx.x % nt;
  int row0 = bm * BM, n0 = bn * BN;
  int tid = threadIdx.x, lane = tid & 31, wvi = tid >> 5;
  int wrow = (wvi >> 2) * 32, wcol = (wvi & 3) * 32;
  int lo = lane & 15, hi = lane >> 4;

  // staging coordinates (loop-invariant)
  int ra = tid >> 2, ca = (tid & 3) * 8;              // A: 8 halfs/thread
  int gra = min(row0 + ra, Mr - 1);                   // clamped row
  int colc0 = (tid * 2) >> 2, kq0 = (tid * 2) & 3;    // B cell 0
  int colc1 = (tid * 2 + 1) >> 2, kq1 = (tid * 2 + 1) & 3;
  int gn0 = min(n0 + colc0, Nc - 1);                  // clamped columns
  int gn1 = min(n0 + colc1, Nc - 1);

  v8f acc[2][2];
#pragma unroll
  for (int i = 0; i < 2; ++i)
#pragma unroll
    for (int j = 0; j < 2; ++j) acc[i][j] = zero8();

  for (int k0 = 0; k0 < Kd; k0 += BK) {
    const _Float16* asrc = A + (size_t)gra * Kd + (k0 + ca);
    copy16_to_lds(asrc, &sA[ra * SA + ca]);
    int k8 = k0 >> 3;
    copy16_to_lds(Bw + ((size_t)(k8 + kq0) * Nc + gn0) * 8,
                  &sBd[colc0 * SBD + kq0 * 4]);
    copy16_to_lds(Bw + ((size_t)(k8 + kq1) * Nc + gn1) * 8,
                  &sBd[colc1 * SBD + kq1 * 4]);
    copy_join();
    __syncthreads();

    HF af[2], bf[2];
#pragma unroll
    for (int i = 0; i < 2; ++i) {
      const _Float16* arow = &sA[(wrow + 16 * i + lo) * SA];
      *(uint4*)&af[i].u[0] = *(const uint4*)(arow + 8 * hi);
      *(uint4*)&af[i].u[4] = *(const uint4*)(arow + 16 + 8 * hi);
    }
#pragma unroll
    for (int j2 = 0; j2 < 2; ++j2) {
      const unsigned int* bcol = &sBd[(wcol + 16 * j2 + lo) * SBD];
      *(uint4*)&bf[j2].u[0] = *(const uint4*)(bcol + 4 * hi);
      *(uint4*)&bf[j2].u[4] = *(const uint4*)(bcol + 8 + 4 * hi);
    }
#pragma unroll
    for (int i = 0; i < 2; ++i)
#pragma unroll
      for (int j2 = 0; j2 < 2; ++j2)
        acc[i][j2] = wmma16(af[i].v, bf[j2].v, acc[i][j2]);
    __syncthreads();
  }

  // epilogue: C layout — VGPR r, lane l: m = r + 8*(l>>4), n = l&15
#pragma unroll
  for (int i = 0; i < 2; ++i)
#pragma unroll
    for (int j2 = 0; j2 < 2; ++j2)
#pragma unroll
      for (int r = 0; r < 8; ++r) {
        int m = row0 + wrow + 16 * i + 8 * hi + r;
        int n = n0 + wcol + 16 * j2 + lo;
        float v = acc[i][j2][r];
        if (bias) v += bias[min(n, Nc - 1)];
        if (act) v = gelu_t(v);
        if (m < Mr && n < Nc) {
          if (Cf) Cf[(size_t)m * Nc + n] = v;
          if (Ch) {
            size_t oi = swzC ? ((((size_t)(m >> 3)) * Nc + n) * 8 +
                                ((m >> 1) & 3) * 2 + (m & 1))
                             : ((size_t)m * Nc + n);
            Ch[oi] = (_Float16)v;
          }
        }
      }
}

// ---------------- fused flash attention (1 wave = 16 q rows, one b,head) ----
// q,k: [BT, Dm] f16 row-major.  v: row-octet interleaved ([BT/8][Dm][8]).
__global__ __launch_bounds__(32) void attn_wmma(
    const _Float16* __restrict__ qh, const _Float16* __restrict__ kh,
    const _Float16* __restrict__ vh, _Float16* __restrict__ oh) {
  __shared__ alignas(16) _Float16 pT[16 * 40];
  const int lane = threadIdx.x;
  const int lo = lane & 15;
  const int hi = lane >> 4;
  int bid = blockIdx.x;
  int qt = bid % nQT;
  int head = (bid / nQT) % Hh;
  int b = bid / (nQT * Hh);
  int qbase = qt * 16;
  const float scale = 0.125f;  // 1/sqrt(DH)

  auto adr = [&](int t, int dh) -> size_t {
    return ((size_t)(b * Tpad + t)) * Dm + head * DH + dh;
  };

  HF aq0, aq1;  // Q fragments for dh 0..31 / 32..63
  {
    int qr = qbase + lo;
    *(uint4*)&aq0.u[0] = *(const uint4*)&qh[adr(qr, 8 * hi)];
    *(uint4*)&aq0.u[4] = *(const uint4*)&qh[adr(qr, 16 + 8 * hi)];
    *(uint4*)&aq1.u[0] = *(const uint4*)&qh[adr(qr, 32 + 8 * hi)];
    *(uint4*)&aq1.u[4] = *(const uint4*)&qh[adr(qr, 48 + 8 * hi)];
  }

  float m_i[8], l_i[8];
  v8f acc[4];
#pragma unroll
  for (int r = 0; r < 8; ++r) { m_i[r] = -3.0e38f; l_i[r] = 0.f; }
#pragma unroll
  for (int c = 0; c < 4; ++c) acc[c] = zero8();

  const uint4* vq = (const uint4*)vh;  // octet cells

  for (int kb = 0; kb < Tpad; kb += 32) {
    // K fragments (B-layout): col = key, contraction = dh (16-B runs)
    HF b0l, b0h, b1l, b1h;
    int key0 = min(kb + lo, Tpad - 1);
    int key1 = min(kb + 16 + lo, Tpad - 1);
    *(uint4*)&b0l.u[0] = *(const uint4*)&kh[adr(key0, 8 * hi)];
    *(uint4*)&b0l.u[4] = *(const uint4*)&kh[adr(key0, 16 + 8 * hi)];
    *(uint4*)&b0h.u[0] = *(const uint4*)&kh[adr(key0, 32 + 8 * hi)];
    *(uint4*)&b0h.u[4] = *(const uint4*)&kh[adr(key0, 48 + 8 * hi)];
    *(uint4*)&b1l.u[0] = *(const uint4*)&kh[adr(key1, 8 * hi)];
    *(uint4*)&b1l.u[4] = *(const uint4*)&kh[adr(key1, 16 + 8 * hi)];
    *(uint4*)&b1h.u[0] = *(const uint4*)&kh[adr(key1, 32 + 8 * hi)];
    *(uint4*)&b1h.u[4] = *(const uint4*)&kh[adr(key1, 48 + 8 * hi)];

    v8f c1 = wmma16(aq0.v, b0l.v, zero8());
    c1 = wmma16(aq1.v, b0h.v, c1);
    v8f c2 = wmma16(aq0.v, b1l.v, zero8());
    c2 = wmma16(aq1.v, b1h.v, c2);

    bool msk1 = (kb + lo) >= Tt;        // per-lane = per-column mask
    bool msk2 = (kb + 16 + lo) >= Tt;
    float alpha[8];
#pragma unroll
    for (int r = 0; r < 8; ++r) {
      float s1 = msk1 ? -1.0e30f : c1[r] * scale;
      float s2 = msk2 ? -1.0e30f : c2[r] * scale;
      float mx = fmaxf(s1, s2);
#pragma unroll
      for (int d = 1; d < 16; d <<= 1) mx = fmaxf(mx, __shfl_xor(mx, d));
      float mn = fmaxf(m_i[r], mx);
      float al = __expf(m_i[r] - mn);
      float p1 = __expf(s1 - mn);
      float p2 = __expf(s2 - mn);
      float sm = p1 + p2;
#pragma unroll
      for (int d = 1; d < 16; d <<= 1) sm += __shfl_xor(sm, d);
      l_i[r] = l_i[r] * al + sm;
      m_i[r] = mn;
      alpha[r] = al;
      // stage P (16x32) to LDS for re-layout into A fragment
      pT[(r + 8 * hi) * 40 + lo]      = (_Float16)p1;
      pT[(r + 8 * hi) * 40 + 16 + lo] = (_Float16)p2;
    }
#pragma unroll
    for (int c = 0; c < 4; ++c)
#pragma unroll
      for (int r = 0; r < 8; ++r) acc[c][r] *= alpha[r];
    __syncthreads();

    HF ap;
    {
      const _Float16* prow = &pT[lo * 40];
      *(uint4*)&ap.u[0] = *(const uint4*)(prow + 8 * hi);
      *(uint4*)&ap.u[4] = *(const uint4*)(prow + 16 + 8 * hi);
    }

    // V fragments: octet cells -> two b128 per fragment
    int o0 = min((kb >> 3) + hi,     OPB - 1);   // clamped (masked keys: P=0)
    int o1 = min((kb >> 3) + 2 + hi, OPB - 1);
#pragma unroll
    for (int c = 0; c < 4; ++c) {
      int dh = head * DH + 16 * c + lo;
      HF bvf;
      *(uint4*)&bvf.u[0] = vq[(size_t)(b * OPB + o0) * Dm + dh];
      *(uint4*)&bvf.u[4] = vq[(size_t)(b * OPB + o1) * Dm + dh];
      acc[c] = wmma16(ap.v, bvf.v, acc[c]);
    }
    __syncthreads();
  }

  float invl[8];
#pragma unroll
  for (int r = 0; r < 8; ++r) invl[r] = 1.0f / l_i[r];
#pragma unroll
  for (int c = 0; c < 4; ++c)
#pragma unroll
    for (int r = 0; r < 8; ++r) {
      int trow = qbase + 8 * hi + r;
      oh[adr(trow, 16 * c + lo)] = (_Float16)(acc[c][r] * invl[r]);
    }
}

// ---------------- residual + LayerNorm (D=512, 1 row per block) ----------------
__global__ __launch_bounds__(256) void ln_kernel(
    const float* __restrict__ x, const float* __restrict__ res,
    const float* __restrict__ w, const float* __restrict__ bb,
    float* __restrict__ outF, _Float16* __restrict__ outH) {
  __shared__ float red[256];
  int row = blockIdx.x, tid = threadIdx.x;
  size_t base = (size_t)row * Dm;
  float v0 = x[base + tid];
  float v1 = x[base + tid + 256];
  if (res) { v0 += res[base + tid]; v1 += res[base + tid + 256]; }
  red[tid] = v0 + v1;
  __syncthreads();
  for (int s = 128; s > 0; s >>= 1) { if (tid < s) red[tid] += red[tid + s]; __syncthreads(); }
  float mu = red[0] * (1.0f / Dm);
  __syncthreads();
  red[tid] = v0 * v0 + v1 * v1;
  __syncthreads();
  for (int s = 128; s > 0; s >>= 1) { if (tid < s) red[tid] += red[tid + s]; __syncthreads(); }
  float var = red[0] * (1.0f / Dm) - mu * mu;
  float rs = rsqrtf(var + 1e-5f);
  float y0 = (v0 - mu) * rs * w[tid] + bb[tid];
  float y1 = (v1 - mu) * rs * w[tid + 256] + bb[tid + 256];
  if (outF) { outF[base + tid] = y0; outF[base + tid + 256] = y1; }
  if (outH) { outH[base + tid] = (_Float16)y0; outH[base + tid + 256] = (_Float16)y1; }
}

// ---------------- retrieval stage (bandwidth bound; VALU) ----------------
__global__ __launch_bounds__(256) void qnorm_kernel(const float* __restrict__ predf,
                                                    float* __restrict__ qn) {
  __shared__ float red[256];
  int b = blockIdx.x, tid = threadIdx.x;
  float s = 0.f;
  for (int i = tid; i < PN; i += 256) {
    int p = i >> 9, n = i & 511;
    float v = predf[((size_t)(b * Tpad + n)) * Pp + p];
    s += v * v;
  }
  red[tid] = s;
  __syncthreads();
  for (int st = 128; st > 0; st >>= 1) { if (tid < st) red[tid] += red[tid + st]; __syncthreads(); }
  if (tid == 0) qn[b] = sqrtf(red[0]);
}

__global__ __launch_bounds__(256) void dots_kernel(
    const float* __restrict__ predf, const float* __restrict__ bank,
    float* __restrict__ dots, float* __restrict__ bnrm) {
  __shared__ float red[256];
  int j = blockIdx.x, tid = threadIdx.x;
  float accq[16];
#pragma unroll
  for (int b = 0; b < 16; ++b) accq[b] = 0.f;
  float accsq = 0.f;
  for (int i = tid; i < PN; i += 256) {
    int p = i >> 9, n = i & 511;
    float bvv;
    if (j < Mm) {
      bvv = bank[(size_t)j * PN + i];
      __builtin_prefetch(&bank[(size_t)j * PN + i + 4096], 0, 1);
    } else {
      bvv = predf[((size_t)((j - Mm) * Tpad + n)) * Pp + p];
    }
    accsq += bvv * bvv;
#pragma unroll
    for (int b = 0; b < 16; ++b)
      accq[b] += bvv * predf[((size_t)(b * Tpad + n)) * Pp + p];
  }
  for (int t = 0; t < 17; ++t) {
    red[tid] = (t < 16) ? accq[t] : accsq;
    __syncthreads();
    for (int s = 128; s > 0; s >>= 1) { if (tid < s) red[tid] += red[tid + s]; __syncthreads(); }
    if (tid == 0) {
      if (t < 16) dots[t * MB + j] = red[0];
      else        bnrm[j] = sqrtf(red[0]);
    }
    __syncthreads();
  }
}

__global__ __launch_bounds__(256) void topk_kernel(
    const float* __restrict__ dots, const float* __restrict__ qn,
    const float* __restrict__ bnrm, int* __restrict__ idx) {
  __shared__ float sims[MB];
  __shared__ float rv[256];
  __shared__ int ri[256];
  int b = blockIdx.x, tid = threadIdx.x;
  float qd = qn[b] + 1e-8f;
  for (int j = tid; j < MB; j += 256)
    sims[j] = dots[b * MB + j] / (qd * (bnrm[j] + 1e-8f));
  __syncthreads();
  for (int k = 0; k < Kk; ++k) {
    float bvv = -3.0e38f;
    int bi = 0;
    for (int j = tid; j < MB; j += 256) {
      float v = sims[j];
      if (v > bvv) { bvv = v; bi = j; }
    }
    rv[tid] = bvv; ri[tid] = bi;
    __syncthreads();
    for (int s = 128; s > 0; s >>= 1) {
      if (tid < s) {
        if (rv[tid + s] > rv[tid] ||
            (rv[tid + s] == rv[tid] && ri[tid + s] < ri[tid])) {
          rv[tid] = rv[tid + s];
          ri[tid] = ri[tid + s];
        }
      }
      __syncthreads();
    }
    if (tid == 0) { idx[b * Kk + k] = ri[0]; sims[ri[0]] = -3.0e38f; }
    __syncthreads();
  }
}

__global__ void fuse_kernel(const float* __restrict__ predf,
                            const float* __restrict__ bank,
                            const int* __restrict__ idx,
                            float* __restrict__ out) {
  int i = blockIdx.x * blockDim.x + threadIdx.x;
  if (i >= Bb * Pp * Nvar) return;
  int n = i & 511;
  int rem = i >> 9;
  int p = rem % Pp;
  int b = rem / Pp;
  float s = 0.f;
#pragma unroll
  for (int k = 0; k < Kk; ++k) {
    int j = idx[b * Kk + k];
    float v;
    if (j < Mm) v = bank[(size_t)j * PN + p * Nvar + n];
    else        v = predf[((size_t)((j - Mm) * Tpad + n)) * Pp + p];
    s += v;
  }
  float pv = predf[((size_t)(b * Tpad + n)) * Pp + p];
  out[i] = 0.5f * pv + 0.5f * (s * (1.0f / Kk));
}

// ---------------- host orchestration ----------------
extern "C" void kernel_launch(void* const* d_in, const int* in_sizes, int n_in,
                              void* d_out, int out_size, void* d_ws, size_t ws_size,
                              hipStream_t stream) {
  (void)in_sizes; (void)n_in; (void)out_size; (void)ws_size;
  const float* x_enc  = (const float*)d_in[0];
  const float* x_mark = (const float*)d_in[1];
  const float* W_emb  = (const float*)d_in[4];
  const float* b_emb  = (const float*)d_in[5];
  const float* Wq = (const float*)d_in[6];
  const float* Wk = (const float*)d_in[7];
  const float* Wv = (const float*)d_in[8];
  const float* Wo = (const float*)d_in[9];
  const float* bq = (const float*)d_in[10];
  const float* bk = (const float*)d_in[11];
  const float* bvb = (const float*)d_in[12];
  const float* bo = (const float*)d_in[13];
  const float* ln1w = (const float*)d_in[14];
  const float* ln1b = (const float*)d_in[15];
  const float* ln2w = (const float*)d_in[16];
  const float* ln2b = (const float*)d_in[17];
  const float* Wf1 = (const float*)d_in[18];
  const float* bf1 = (const float*)d_in[19];
  const float* Wf2 = (const float*)d_in[20];
  const float* bf2 = (const float*)d_in[21];
  const float* lnfw = (const float*)d_in[22];
  const float* lnfb = (const float*)d_in[23];
  const float* Wp = (const float*)d_in[24];
  const float* bp = (const float*)d_in[25];
  const float* bank = (const float*)d_in[26];
  float* out = (float*)d_out;

  char* wsb = (char*)d_ws;
  size_t off = 0;
  auto alloc = [&](size_t bytes) -> void* {
    void* p = (void*)(wsb + off);
    off += (bytes + 255) & ~(size_t)255;
    return p;
  };
  _Float16* tok_h = (_Float16*)alloc((size_t)BT * Lctx * 2);
  float*    h_f   = (float*)   alloc((size_t)BT * Dm * 4);
  _Float16* h_h   = (_Float16*)alloc((size_t)BT * Dm * 2);
  _Float16* q_h   = (_Float16*)alloc((size_t)BT * Dm * 2);
  _Float16* k_h   = (_Float16*)alloc((size_t)BT * Dm * 2);
  _Float16* v_h   = (_Float16*)alloc((size_t)BT * Dm * 2);   // octet-swizzled
  _Float16* o_h   = (_Float16*)alloc((size_t)BT * Dm * 2);
  float*    scr_f = (float*)   alloc((size_t)BT * Dm * 4);
  _Float16* y1_h  = (_Float16*)alloc((size_t)BT * DFF * 2);
  float*    predf = (float*)   alloc((size_t)BT * Pp * 4);
  _Float16* wemb_h = (_Float16*)alloc((size_t)Lctx * Dm * 2);
  _Float16* wq_h   = (_Float16*)alloc((size_t)Dm * Dm * 2);
  _Float16* wk_h   = (_Float16*)alloc((size_t)Dm * Dm * 2);
  _Float16* wv_h   = (_Float16*)alloc((size_t)Dm * Dm * 2);
  _Float16* wo_h   = (_Float16*)alloc((size_t)Dm * Dm * 2);
  _Float16* wf1_h  = (_Float16*)alloc((size_t)Dm * DFF * 2);
  _Float16* wf2_h  = (_Float16*)alloc((size_t)DFF * Dm * 2);
  _Float16* wp_h   = (_Float16*)alloc((size_t)Dm * Pp * 2);
  float* dotsb = (float*)alloc((size_t)Bb * MB * 4);
  float* qn    = (float*)alloc(Bb * 4);
  float* bnrm  = (float*)alloc(MB * 4);
  int*   idx   = (int*)alloc(Bb * Kk * 4);

  auto cvt = [&](const float* s, _Float16* d, int Kd, int Nc) {
    cvt_f32_f16_oct<<<(Kd * Nc + 255) / 256, 256, 0, stream>>>(s, d, Kd, Nc);
  };
  auto gemm = [&](const _Float16* A, const _Float16* Bm, const float* bias,
                  float* Cf, _Float16* Ch, int Mr, int Kd, int Nc, int act,
                  int swzC) {
    int nt = (Nc + BN - 1) / BN, mt = (Mr + BM - 1) / BM;
    gemm_wmma<<<mt * nt, 256, 0, stream>>>(A, Bm, bias, Cf, Ch, Mr, Kd, Nc,
                                           act, swzC);
  };

  pack_tok<<<(BT * Lctx + 255) / 256, 256, 0, stream>>>(x_enc, x_mark, tok_h);
  cvt(W_emb, wemb_h, Lctx, Dm);
  gemm(tok_h, wemb_h, b_emb, h_f, h_h, BT, Lctx, Dm, 0, 0);

  for (int il = 0; il < NLay; ++il) {
    cvt(Wq + (size_t)il * Dm * Dm, wq_h, Dm, Dm);
    cvt(Wk + (size_t)il * Dm * Dm, wk_h, Dm, Dm);
    cvt(Wv + (size_t)il * Dm * Dm, wv_h, Dm, Dm);
    cvt(Wo + (size_t)il * Dm * Dm, wo_h, Dm, Dm);
    cvt(Wf1 + (size_t)il * Dm * DFF, wf1_h, Dm, DFF);
    cvt(Wf2 + (size_t)il * DFF * Dm, wf2_h, DFF, Dm);

    gemm(h_h, wq_h, bq + il * Dm, nullptr, q_h, BT, Dm, Dm, 0, 0);
    gemm(h_h, wk_h, bk + il * Dm, nullptr, k_h, BT, Dm, Dm, 0, 0);
    gemm(h_h, wv_h, bvb + il * Dm, nullptr, v_h, BT, Dm, Dm, 0, 1);  // octet V
    attn_wmma<<<Bb * Hh * nQT, 32, 0, stream>>>(q_h, k_h, v_h, o_h);
    gemm(o_h, wo_h, bo + il * Dm, scr_f, nullptr, BT, Dm, Dm, 0, 0);
    ln_kernel<<<BT, 256, 0, stream>>>(scr_f, h_f, ln1w + il * Dm, ln1b + il * Dm, h_f, h_h);
    gemm(h_h, wf1_h, bf1 + il * DFF, nullptr, y1_h, BT, Dm, DFF, 1, 0);
    gemm(y1_h, wf2_h, bf2 + il * Dm, scr_f, nullptr, BT, DFF, Dm, 0, 0);
    ln_kernel<<<BT, 256, 0, stream>>>(scr_f, h_f, ln2w + il * Dm, ln2b + il * Dm, h_f, h_h);
  }

  ln_kernel<<<BT, 256, 0, stream>>>(h_f, nullptr, lnfw, lnfb, h_f, h_h);
  cvt(Wp, wp_h, Dm, Pp);
  gemm(h_h, wp_h, bp, predf, nullptr, BT, Dm, Pp, 0, 0);

  qnorm_kernel<<<Bb, 256, 0, stream>>>(predf, qn);
  dots_kernel<<<MB, 256, 0, stream>>>(predf, bank, dotsb, bnrm);
  topk_kernel<<<Bb, 256, 0, stream>>>(dotsb, qn, bnrm, idx);
  fuse_kernel<<<(Bb * Pp * Nvar + 255) / 256, 256, 0, stream>>>(predf, bank, idx, out);
}
